// VectorQuantizer_24240795419231
// MI455X (gfx1250) — matrix operand order repaired and checked by reading, not compiled, not run on hardware
//
#include <hip/hip_runtime.h>
#include <hip/hip_bf16.h>

// ---------------------------------------------------------------------------
// VQ nearest-codebook search on gfx1250.
//   distances = ||x||^2 + ||e||^2 - 2 x.e ; argmin over K ; gather.
//   ||x||^2 is row-constant -> dropped. dist = e_norm[k] - 2*dot(x,e_k).
// Matmul via v_wmma_f32_16x16x32_bf16; codebook tiles staged to LDS with
// double-buffered global_load_async_to_lds_b128 (ASYNCcnt), pre-swizzled
// into the WMMA B-fragment layout so compute reads are 2x ds_load_b128.
// ---------------------------------------------------------------------------

typedef __attribute__((ext_vector_type(16))) __bf16 v16bf;
typedef __attribute__((ext_vector_type(8)))  float  v8f;

union BFu {
    v16bf        v;
    unsigned int u[8];
    uint4        q[2];
};

#define DIM   256
#define KMAX  1024
#define TILEB 8192          // one B tile: 16 codes x 256 bf16 = 8 KB

static __device__ inline unsigned short f2bf(float f) {
    unsigned int u = __float_as_uint(f);
    u += 0x7FFFu + ((u >> 16) & 1u);       // round to nearest even
    return (unsigned short)(u >> 16);
}
static __device__ inline unsigned int pack2(float a, float b) {
    return (unsigned int)f2bf(a) | ((unsigned int)f2bf(b) << 16);
}

// async 16-byte copy global -> LDS (ASYNCcnt-tracked, bypasses VGPR data path)
static __device__ inline void async_copy_b128(unsigned lds_off, const char* gsrc) {
    asm volatile("global_load_async_to_lds_b128 %0, %1, off"
                 :: "v"(lds_off), "v"(gsrc) : "memory");
}
static __device__ inline void wait_asynccnt0() {
    asm volatile("s_wait_asynccnt 0x0" ::: "memory");
}

// ---------------------------------------------------------------------------
// Kernel 1: per-code squared norms + bf16 copy of the codebook (one wave/row).
// ---------------------------------------------------------------------------
__global__ __launch_bounds__(32)
void vq_prep_kernel(const float* __restrict__ ew,
                    unsigned short* __restrict__ ebf,
                    float* __restrict__ enorm) {
    const int code = blockIdx.x;
    const int lane = threadIdx.x;          // 0..31, wave32
    const float* row = ew + (size_t)code * DIM;
    float s = 0.0f;
#pragma unroll
    for (int i = 0; i < DIM / 32; ++i) {
        float v = row[lane + i * 32];
        s += v * v;
        ebf[(size_t)code * DIM + lane + i * 32] = f2bf(v);
    }
#pragma unroll
    for (int m = 16; m >= 1; m >>= 1) s += __shfl_xor(s, m, 32);
    if (lane == 0) enorm[code] = s;
}

// ---------------------------------------------------------------------------
// Kernel 2: main search. 256 threads = 8 waves; each wave owns 16 rows.
// ---------------------------------------------------------------------------
__global__ __launch_bounds__(256)
void vq_main_kernel(const float* __restrict__ x,
                    const float* __restrict__ ew,        // f32 codebook (gather)
                    const char* __restrict__ ebf_bytes,  // bf16 codebook, row-major
                    const float* __restrict__ enorm,
                    float* __restrict__ outq,
                    int* __restrict__ outi,
                    int nrows, int K) {
    const int tid  = threadIdx.x;
    const int lane = tid & 31;
    const int wave = tid >> 5;
    const int half = lane >> 4;            // 0: lanes 0-15, 1: lanes 16-31
    const int l15  = lane & 15;
    const int rowBase = blockIdx.x * 128 + wave * 16;

    // LDS: double-buffered swizzled B tiles + norm table + winner indices
    __shared__ uint4 sB4[2 * TILEB / 16];  // 16 KB, 16B-aligned
    __shared__ float sNorm[KMAX];          // 4 KB
    __shared__ int   sIdx[128];

    char* sB = (char*)sB4;
    const unsigned sB_lds = (unsigned)(size_t)sB;   // LDS byte offset of sB

    // ---- stage norm table into LDS ----------------------------------------
    for (int i = tid; i < K && i < KMAX; i += 256) sNorm[i] = enorm[i];

    // ---- Preload all A fragments (16 rows x 256 dims = 8 chunks of K=32) ---
    int arow = rowBase + l15;
    if (arow >= nrows) arow = nrows - 1;   // clamp: keep EXEC uniform for WMMA
    const float* xr = x + (size_t)arow * DIM;

    BFu A[8];
#pragma unroll
    for (int dc = 0; dc < 8; ++dc) {
        const int d0 = dc * 32 + half * 8;          // K = half*8 + 0..7
        float4 fa = *(const float4*)(xr + d0);
        float4 fb = *(const float4*)(xr + d0 + 4);
        float4 fc = *(const float4*)(xr + d0 + 16); // K = 16 + half*8 + 0..7
        float4 fd = *(const float4*)(xr + d0 + 20);
        A[dc].u[0] = pack2(fa.x, fa.y);
        A[dc].u[1] = pack2(fa.z, fa.w);
        A[dc].u[2] = pack2(fb.x, fb.y);
        A[dc].u[3] = pack2(fb.z, fb.w);
        A[dc].u[4] = pack2(fc.x, fc.y);
        A[dc].u[5] = pack2(fc.z, fc.w);
        A[dc].u[6] = pack2(fd.x, fd.y);
        A[dc].u[7] = pack2(fd.z, fd.w);
    }

    // ---- async prefetch of one 8 KB B tile, pre-swizzled to fragment layout.
    // Swizzled layout: [dc 0..7][lane' 0..31][16B x 2]; lane' reads its own
    // 32 contiguous bytes per chunk -> conflict-free ds_load_b128 pairs.
    // 256 threads x 2 units x 16 B = 8 KB.
    auto prefetch_tile = [&](int ct, int buf) {
#pragma unroll
        for (int p = 0; p < 2; ++p) {
            const int j      = tid + p * 256;   // 16B unit id, 0..511
            const int off16  = j & 1;
            const int lanep  = (j >> 1) & 31;
            const int dc     = j >> 6;
            const unsigned dst = sB_lds + (unsigned)(buf * TILEB + dc * 1024 +
                                                     lanep * 32 + off16 * 16);
            const char* src = ebf_bytes +
                ((size_t)(ct * 16 + (lanep & 15)) * (DIM * 2) +  // code row
                 dc * 64 + (lanep >> 4) * 16 + off16 * 32);      // K chunk
            async_copy_b128(dst, src);
        }
    };

    prefetch_tile(0, 0);

    float bestd[8];
    int   besti[8];
#pragma unroll
    for (int j = 0; j < 8; ++j) { bestd[j] = 3.4e38f; besti[j] = 0; }

    // ---- Sweep code tiles: 16 codes per tile, D=256 in 8 WMMA steps --------
    const int ntiles = K >> 4;
    for (int ct = 0; ct < ntiles; ++ct) {
        const int buf = ct & 1;

        wait_asynccnt0();                  // my async stores to LDS landed
        __syncthreads();                   // everyone's landed; prior reads done

        if (ct + 1 < ntiles) prefetch_tile(ct + 1, buf ^ 1);

        const uint4* bt = (const uint4*)(sB + buf * TILEB + lane * 32);

        v8f acc = {0.f, 0.f, 0.f, 0.f, 0.f, 0.f, 0.f, 0.f};
#pragma unroll
        for (int dc = 0; dc < 8; ++dc) {
            BFu B;
            B.q[0] = bt[dc * 64];          // ds_load_b128
            B.q[1] = bt[dc * 64 + 1];      // ds_load_b128
            acc = __builtin_amdgcn_wmma_f32_16x16x32_bf16(
                false, A[dc].v, false, B.v, (short)0, acc, false, false);
        }

        const int   code = (ct << 4) + l15;   // this lane's column
        const float en   = sNorm[code];
#pragma unroll
        for (int j = 0; j < 8; ++j) {
            float d = en - 2.0f * acc[j];     // ||e||^2 - 2 x.e
            if (d < bestd[j]) { bestd[j] = d; besti[j] = code; }
        }
    }

    // ---- Argmin across the 16 lanes of each column group -------------------
#pragma unroll
    for (int j = 0; j < 8; ++j) {
        float d = bestd[j];
        int   i = besti[j];
#pragma unroll
        for (int m = 8; m >= 1; m >>= 1) {    // xor masks stay inside half-group
            float od = __shfl_xor(d, m, 32);
            int   oi = __shfl_xor(i, m, 32);
            if (od < d || (od == d && oi < i)) { d = od; i = oi; }
        }
        besti[j] = i;
    }
    // C layout: VGPR j <-> row (half*8 + j); winner lives in column-lane 0
    if (l15 == 0) {
#pragma unroll
        for (int j = 0; j < 8; ++j) sIdx[wave * 16 + half * 8 + j] = besti[j];
    }
    __syncthreads();

    // ---- Gather quantized rows + write indices -----------------------------
    for (int r = 0; r < 16; ++r) {
        const int row = rowBase + r;
        if (row >= nrows) break;
        const int idx = sIdx[wave * 16 + r];
        const float4* src = (const float4*)(ew   + (size_t)idx * DIM);
        float4*       dst = (float4*)      (outq + (size_t)row * DIM);
        dst[lane * 2]     = src[lane * 2];     // 32 lanes x 2 float4 = 256 f32
        dst[lane * 2 + 1] = src[lane * 2 + 1];
        if (lane == 0) outi[row] = idx;
    }
}

// ---------------------------------------------------------------------------
extern "C" void kernel_launch(void* const* d_in, const int* in_sizes, int n_in,
                              void* d_out, int out_size, void* d_ws, size_t ws_size,
                              hipStream_t stream) {
    const float* x  = (const float*)d_in[0];   // [N, 256] flat
    const float* ew = (const float*)d_in[1];   // [K, 256]
    const int nrows = in_sizes[0] / DIM;       // 65536
    const int K     = in_sizes[1] / DIM;       // 1024

    // workspace: bf16 codebook (K*256*2 B) then f32 norms (K*4 B)  ~516 KB
    unsigned short* ebf   = (unsigned short*)d_ws;
    float*          enorm = (float*)((char*)d_ws + (size_t)K * DIM * 2);

    float* outq = (float*)d_out;
    int*   outi = (int*)(outq + (size_t)nrows * DIM);  // indices after quantized

    vq_prep_kernel<<<K, 32, 0, stream>>>(ew, ebf, enorm);

    const int blocks = (nrows + 127) / 128;    // 128 rows per block (8 waves x 16)
    vq_main_kernel<<<blocks, 256, 0, stream>>>(x, ew, (const char*)ebf,
                                               enorm, outq, outi, nrows, K);
}
